// Speller_33260226740248
// MI455X (gfx1250) — matrix operand order repaired and checked by reading, not compiled, not run on hardware
//
#include <hip/hip_runtime.h>
#include <hip/hip_bf16.h>
#include <math.h>

// Problem dims (match reference)
#define B_   64
#define U_   1024
#define K_   512
#define P_   256
#define C_   64
#define T_   77
#define G4K  2048     // 4*K
#define XDIM 1088     // (C+K) + K = 576 + 512

typedef __attribute__((ext_vector_type(16))) __bf16 v16bf;
typedef __attribute__((ext_vector_type(8)))  __bf16 v8bf;
typedef __attribute__((ext_vector_type(8)))  float  v8f;

// ---------------- wave reductions (wave32) ----------------
__device__ inline float wave_max(float v) {
#pragma unroll
  for (int o = 16; o > 0; o >>= 1) v = fmaxf(v, __shfl_xor(v, o, 32));
  return v;
}
__device__ inline float wave_sum(float v) {
#pragma unroll
  for (int o = 16; o > 0; o >>= 1) v += __shfl_xor(v, o, 32);
  return v;
}

// ---------------- f32 -> bf16 converters ----------------
__global__ void cvt_f32_bf16(const float* __restrict__ s, __bf16* __restrict__ d, int n) {
  for (int i = blockIdx.x * blockDim.x + threadIdx.x; i < n; i += gridDim.x * blockDim.x)
    d[i] = (__bf16)s[i];
}

// Pack [W_ih (2048x576) | W_hh (2048x512)] -> Wcomb (2048x1088) bf16, row-major
__global__ void pack_wcomb(const float* __restrict__ W_ih, const float* __restrict__ W_hh,
                           __bf16* __restrict__ Wc) {
  const int n = G4K * XDIM;
  for (int i = blockIdx.x * blockDim.x + threadIdx.x; i < n; i += gridDim.x * blockDim.x) {
    int g = i / XDIM, j = i - g * XDIM;
    float v = (j < (C_ + K_)) ? W_ih[(size_t)g * (C_ + K_) + j]
                              : W_hh[(size_t)g * K_ + (j - (C_ + K_))];
    Wc[i] = (__bf16)v;
  }
}

// ---------------- hp = relu(h * psi^T + psi_b) ----------------
// WMMA bf16, A converted f32->bf16 on the fly, software-pipelined (double-buffered frags).
// A frag (16x32): lane=hi*16+lo; row m0+lo; elems 0..15 <- k+hi*16 (contiguous).
// B frag (32x16): lane column n0+lo; 16 contiguous bf16 at k+hi*16.
// D: reg r -> row m0+hi*8+r, col n0+lo.
__global__ void hp_gemm(const float* __restrict__ h,      // (B*U) x K f32 row-major
                        const __bf16* __restrict__ Wpsi,  // P x K bf16
                        const float* __restrict__ bias,   // P
                        __bf16* __restrict__ hp,          // (B*U) x P bf16
                        int M, int N, int Kd) {
  const int NT = 4;
  int wave = (blockIdx.x * blockDim.x + threadIdx.x) >> 5;
  int lane = threadIdx.x & 31;
  int lo = lane & 15, hi = lane >> 4;
  int ngroups = N / (16 * NT);
  int mt = wave / ngroups;
  int ng = wave - mt * ngroups;
  if (mt * 16 >= M) return;

  v8f acc[NT];
#pragma unroll
  for (int t = 0; t < NT; t++)
#pragma unroll
    for (int r = 0; r < 8; r++) acc[t][r] = 0.f;

  const float* xrow = h + (size_t)(mt * 16 + lo) * Kd;
  const __bf16* wrow[NT];
#pragma unroll
  for (int t = 0; t < NT; t++)
    wrow[t] = Wpsi + (size_t)((ng * NT + t) * 16 + lo) * Kd;

  v16bf a0, a1, b0[NT], b1[NT];
#pragma unroll
  for (int j = 0; j < 8; j++) a0[j] = (__bf16)xrow[hi * 16 + j];
#pragma unroll
  for (int j = 0; j < 8; j++) a0[8 + j] = (__bf16)xrow[hi * 16 + 8 + j];
#pragma unroll
  for (int t = 0; t < NT; t++) b0[t] = *(const v16bf*)(wrow[t] + hi * 16);

  for (int k = 32; k < Kd; k += 32) {
    // prefetch next iteration's fragments while this one's WMMAs run
#pragma unroll
    for (int j = 0; j < 8; j++) a1[j] = (__bf16)xrow[k + hi * 16 + j];
#pragma unroll
    for (int j = 0; j < 8; j++) a1[8 + j] = (__bf16)xrow[k + hi * 16 + 8 + j];
#pragma unroll
    for (int t = 0; t < NT; t++) b1[t] = *(const v16bf*)(wrow[t] + k + hi * 16);
#pragma unroll
    for (int t = 0; t < NT; t++)
      acc[t] = __builtin_amdgcn_wmma_f32_16x16x32_bf16(false, a0, false, b0[t],
                                                       (short)0, acc[t], false, false);
    a0 = a1;
#pragma unroll
    for (int t = 0; t < NT; t++) b0[t] = b1[t];
  }
#pragma unroll
  for (int t = 0; t < NT; t++)
    acc[t] = __builtin_amdgcn_wmma_f32_16x16x32_bf16(false, a0, false, b0[t],
                                                     (short)0, acc[t], false, false);

#pragma unroll
  for (int t = 0; t < NT; t++) {
    int n = (ng * NT + t) * 16 + lo;
    float bv = bias[n];
#pragma unroll
    for (int r = 0; r < 8; r++) {
      int m = mt * 16 + hi * 8 + r;
      float v = acc[t][r] + bv;
      hp[(size_t)m * N + n] = (__bf16)(v > 0.f ? v : 0.f);
    }
  }
}

// ---------------- init decode state ----------------
// Xc0[b] = [onehot(class0) (64) | h[b,0,:] (512) | hs0=0 (512)], cs = 0
__global__ void init_state(const float* __restrict__ h, float* __restrict__ cs,
                           __bf16* __restrict__ Xc0) {
  int idx = blockIdx.x * blockDim.x + threadIdx.x;
  if (idx >= B_ * XDIM) return;
  int b = idx / XDIM, j = idx - b * XDIM;
  __bf16 v;
  if (j < C_)            v = (__bf16)(j == 0 ? 1.f : 0.f);
  else if (j < C_ + K_)  v = (__bf16)h[(size_t)b * U_ * K_ + (j - C_)];
  else                   v = (__bf16)0.f;
  Xc0[idx] = v;
  if (j < K_) cs[(size_t)b * K_ + j] = 0.f;
}

// ---------------- fused gates GEMM (WMMA) + LSTM nonlinearity ----------------
// Grid: 32 blocks (16-wide kk column groups), 512 threads = 16 waves.
// Wave w: mt = w&3 (M rows), sec = w>>2 (gate section i/f/g/o).
// Block owns gate columns {kk, 512+kk, 1024+kk, 1536+kk} for kk in [ng*16, ng*16+16),
// for ALL 64 batch rows -> activations computable in-block after syncthreads.
__global__ void gates_lstm(const __bf16* __restrict__ Xcur,   // 64 x 1088 bf16 (step input)
                           const __bf16* __restrict__ Wcomb,  // 2048 x 1088 bf16
                           const float* __restrict__ b_ih, const float* __restrict__ b_hh,
                           float* __restrict__ cs,            // 64 x 512 (in/out)
                           __bf16* __restrict__ Xnext,        // 64 x 1088 (hs slice written)
                           __bf16* __restrict__ hs_bf) {      // 64 x 512
  __shared__ float s_g[4][B_][16];
  int tid = threadIdx.x;
  int wave = tid >> 5, lane = tid & 31;
  int lo = lane & 15, hi = lane >> 4;
  int mt = wave & 3, sec = wave >> 2;
  int ng = blockIdx.x;
  int n0 = sec * K_ + ng * 16;

  v8f acc;
#pragma unroll
  for (int r = 0; r < 8; r++) acc[r] = 0.f;

  const __bf16* xrow = Xcur + (size_t)(mt * 16 + lo) * XDIM;
  const __bf16* wrow = Wcomb + (size_t)(n0 + lo) * XDIM;

  v16bf a0, a1, b0, b1;
  *(v8bf*)&a0       = *(const v8bf*)(xrow + hi * 16);
  *((v8bf*)&a0 + 1) = *(const v8bf*)(xrow + hi * 16 + 8);
  b0 = *(const v16bf*)(wrow + hi * 16);

  for (int k = 32; k < XDIM; k += 32) {
    if (k + 128 < XDIM) __builtin_prefetch(wrow + k + 128, 0, 0);
    *(v8bf*)&a1       = *(const v8bf*)(xrow + k + hi * 16);
    *((v8bf*)&a1 + 1) = *(const v8bf*)(xrow + k + hi * 16 + 8);
    b1 = *(const v16bf*)(wrow + k + hi * 16);
    acc = __builtin_amdgcn_wmma_f32_16x16x32_bf16(false, a0, false, b0,
                                                  (short)0, acc, false, false);
    a0 = a1;
    b0 = b1;
  }
  acc = __builtin_amdgcn_wmma_f32_16x16x32_bf16(false, a0, false, b0,
                                                (short)0, acc, false, false);

#pragma unroll
  for (int r = 0; r < 8; r++) s_g[sec][mt * 16 + hi * 8 + r][lo] = acc[r];
  __syncthreads();

  // activation: 64 b x 16 cols = 1024 items over 512 threads
  for (int it = tid; it < B_ * 16; it += 512) {
    int b = it >> 4, col = it & 15;
    int kk = ng * 16 + col;
    float gi = s_g[0][b][col] + b_ih[kk]            + b_hh[kk];
    float gf = s_g[1][b][col] + b_ih[K_ + kk]       + b_hh[K_ + kk];
    float gg = s_g[2][b][col] + b_ih[2 * K_ + kk]   + b_hh[2 * K_ + kk];
    float go = s_g[3][b][col] + b_ih[3 * K_ + kk]   + b_hh[3 * K_ + kk];
    float si = 1.f / (1.f + expf(-gi));
    float sf = 1.f / (1.f + expf(-gf));
    float so = 1.f / (1.f + expf(-go));
    int ci = b * K_ + kk;
    float cn = sf * cs[ci] + si * tanhf(gg);
    float hn = so * tanhf(cn);
    cs[ci] = cn;
    __bf16 hb = (__bf16)hn;
    hs_bf[ci] = hb;
    Xnext[(size_t)b * XDIM + (C_ + K_) + kk] = hb;
  }
}

// ---------------- fused: si -> attention (e/softmax/ctx) -> logits -> log_softmax ----------------
// One block per batch row b, 512 threads (16 waves) for memory-level parallelism on the
// L2-streaming phases (e reads hp[b] 0.5MB, ctx reads h[b] 2MB). All intermediates in LDS.
__global__ void attn_fused(const float* __restrict__ h,      // B x U x K f32
                           const __bf16* __restrict__ hp,    // B x U x P bf16
                           const __bf16* __restrict__ hs_bf, // B x K bf16
                           const __bf16* __restrict__ phi,   // P x K bf16
                           const float* __restrict__ phi_b,  // P
                           const __bf16* __restrict__ cd,    // C x 2K bf16
                           const float* __restrict__ cd_b,   // C
                           const int* __restrict__ y,        // B x T
                           float* __restrict__ logps,        // T x B x C
                           float* __restrict__ atts,         // T x B x U
                           __bf16* __restrict__ Xnext, int t) {
  __shared__ float s_hs[K_];
  __shared__ float s_si[P_];
  __shared__ float s_e[U_];
  __shared__ float s_ctx[K_];
  __shared__ float s_log[C_];
  __shared__ float s_red[64];
  const int NTH = 512;
  int b = blockIdx.x, tid = threadIdx.x;

  // hs row -> LDS (f32)
  if (tid < K_) s_hs[tid] = (float)hs_bf[(size_t)b * K_ + tid];
  __syncthreads();

  // si[p] = relu(hs . phi[p] + phi_b[p]); one p per thread (first 256 threads)
  if (tid < P_) {
    const __bf16* prow = phi + (size_t)tid * K_;
    float acc = 0.f;
#pragma unroll 2
    for (int k = 0; k < K_; k += 8) {
      v8bf pv = *(const v8bf*)(prow + k);
#pragma unroll
      for (int j = 0; j < 8; j++) acc += s_hs[k + j] * (float)pv[j];
    }
    acc += phi_b[tid];
    s_si[tid] = acc > 0.f ? acc : 0.f;
  }
  __syncthreads();

  // e[u] = si . hp[b,u,:]  (2 u-rows per thread)
  const __bf16* hpb = hp + (size_t)b * U_ * P_;
  for (int u = tid; u < U_; u += NTH) {
    const __bf16* row = hpb + (size_t)u * P_;
    float acc = 0.f;
#pragma unroll 4
    for (int p = 0; p < P_; p += 8) {
      v8bf hv = *(const v8bf*)(row + p);
#pragma unroll
      for (int j = 0; j < 8; j++) acc += s_si[p + j] * (float)hv[j];
    }
    s_e[u] = acc;
  }
  __syncthreads();

  // softmax over U (16 waves)
  float m = -INFINITY;
  for (int u = tid; u < U_; u += NTH) m = fmaxf(m, s_e[u]);
  m = wave_max(m);
  if ((tid & 31) == 0) s_red[tid >> 5] = m;
  __syncthreads();
  float M = s_red[0];
#pragma unroll
  for (int i = 1; i < 16; i++) M = fmaxf(M, s_red[i]);
  __syncthreads();

  float s = 0.f;
  for (int u = tid; u < U_; u += NTH) {
    float ex = expf(s_e[u] - M);
    s_e[u] = ex;
    s += ex;
  }
  s = wave_sum(s);
  if ((tid & 31) == 0) s_red[tid >> 5] = s;
  __syncthreads();
  float S = 0.f;
#pragma unroll
  for (int i = 0; i < 16; i++) S += s_red[i];
  float inv = 1.f / S;

  float* arow = atts + ((size_t)t * B_ + b) * U_;
  for (int u = tid; u < U_; u += NTH) {
    float a = s_e[u] * inv;
    s_e[u] = a;
    arow[u] = a;
  }
  __syncthreads();

  // ctx[k] = sum_u a[u] * h[b,u,k]  (raw f32 h, as in reference; one column per thread,
  // per-u rows of 512 floats are fully coalesced across the block)
  const float* hb = h + (size_t)b * U_ * K_;
  if (tid < K_) {
    int k = tid;
    float acc = 0.f;
#pragma unroll 4
    for (int u = 0; u < U_; u++) acc += s_e[u] * hb[(size_t)u * K_ + k];
    s_ctx[k] = acc;
    Xnext[(size_t)b * XDIM + C_ + k] = (__bf16)acc;   // next-step x = [y_oh_t, ctx]
  }
  // one-hot of current y_t for next step
  if (tid < C_) {
    int cls = y[(size_t)b * T_ + t];
    Xnext[(size_t)b * XDIM + tid] = (__bf16)(tid == cls ? 1.f : 0.f);
  }
  __syncthreads();

  // logits[c] = [hs|ctx] . cd[c] + cd_b[c]; 8 threads per class, 128 elems each
  {
    int c = tid >> 3, q = tid & 7;
    const __bf16* crow = cd + (size_t)c * (2 * K_) + q * 128;
    const float* src = (q < 4) ? s_hs : s_ctx;
    int base = (q < 4) ? q * 128 : (q - 4) * 128;
    float acc = 0.f;
#pragma unroll 2
    for (int k = 0; k < 128; k += 8) {
      v8bf cv = *(const v8bf*)(crow + k);
#pragma unroll
      for (int j = 0; j < 8; j++) acc += src[base + k + j] * (float)cv[j];
    }
    acc += __shfl_xor(acc, 1, 32);
    acc += __shfl_xor(acc, 2, 32);
    acc += __shfl_xor(acc, 4, 32);
    if (q == 0) s_log[c] = acc + cd_b[c];
  }
  __syncthreads();

  // log-softmax over C=64
  if (tid < C_) s_red[tid] = s_log[tid];
  __syncthreads();
  for (int st = 32; st >= 1; st >>= 1) {
    if (tid < st) s_red[tid] = fmaxf(s_red[tid], s_red[tid + st]);
    __syncthreads();
  }
  float Ml = s_red[0];
  __syncthreads();
  if (tid < C_) s_red[tid] = expf(s_log[tid] - Ml);
  __syncthreads();
  for (int st = 32; st >= 1; st >>= 1) {
    if (tid < st) s_red[tid] += s_red[tid + st];
    __syncthreads();
  }
  float lse = logf(s_red[0]);
  if (tid < C_) logps[((size_t)t * B_ + b) * C_ + tid] = s_log[tid] - Ml - lse;
}

// ---------------- host ----------------
static inline size_t align256(size_t x) { return (x + 255) & ~(size_t)255; }

extern "C" void kernel_launch(void* const* d_in, const int* in_sizes, int n_in,
                              void* d_out, int out_size, void* d_ws, size_t ws_size,
                              hipStream_t stream) {
  const float* h     = (const float*)d_in[0];
  const int*   y     = (const int*)d_in[1];
  const float* W_ih  = (const float*)d_in[2];
  const float* W_hh  = (const float*)d_in[3];
  const float* b_ih  = (const float*)d_in[4];
  const float* b_hh  = (const float*)d_in[5];
  const float* phi_w = (const float*)d_in[6];
  const float* phi_b = (const float*)d_in[7];
  const float* psi_w = (const float*)d_in[8];
  const float* psi_b = (const float*)d_in[9];
  const float* cd_w  = (const float*)d_in[10];
  const float* cd_b  = (const float*)d_in[11];

  char* ws = (char*)d_ws;
  size_t off = 0;
  __bf16* hp_bf  = (__bf16*)(ws + off); off = align256(off + (size_t)B_ * U_ * P_ * 2);
  __bf16* Wcomb  = (__bf16*)(ws + off); off = align256(off + (size_t)G4K * XDIM * 2);
  __bf16* phi_bf = (__bf16*)(ws + off); off = align256(off + (size_t)P_ * K_ * 2);
  __bf16* psi_bf = (__bf16*)(ws + off); off = align256(off + (size_t)P_ * K_ * 2);
  __bf16* cd_bf  = (__bf16*)(ws + off); off = align256(off + (size_t)C_ * 2 * K_ * 2);
  __bf16* Xc0    = (__bf16*)(ws + off); off = align256(off + (size_t)B_ * XDIM * 2);
  __bf16* Xc1    = (__bf16*)(ws + off); off = align256(off + (size_t)B_ * XDIM * 2);
  __bf16* hs_bf  = (__bf16*)(ws + off); off = align256(off + (size_t)B_ * K_ * 2);
  float*  cs     = (float*)(ws + off);  off = align256(off + (size_t)B_ * K_ * 4);
  (void)ws_size;

  float* logps = (float*)d_out;                    // T x B x C
  float* atts  = logps + (size_t)T_ * B_ * C_;     // T x B x U

  // ---- prep: bf16 weights ----
  pack_wcomb<<<(G4K * XDIM + 255) / 256, 256, 0, stream>>>(W_ih, W_hh, Wcomb);
  cvt_f32_bf16<<<256, 256, 0, stream>>>(phi_w, phi_bf, P_ * K_);
  cvt_f32_bf16<<<256, 256, 0, stream>>>(psi_w, psi_bf, P_ * K_);
  cvt_f32_bf16<<<256, 256, 0, stream>>>(cd_w, cd_bf, C_ * 2 * K_);

  // ---- hp = relu(h @ psi^T + psi_b), M = B*U ----
  {
    int waves = ((B_ * U_) / 16) * (P_ / 64);
    hp_gemm<<<(waves * 32 + 255) / 256, 256, 0, stream>>>(h, psi_bf, psi_b, hp_bf,
                                                          B_ * U_, P_, K_);
  }

  // ---- init state ----
  init_state<<<(B_ * XDIM + 255) / 256, 256, 0, stream>>>(h, cs, Xc0);

  // ---- teacher-forced decode: 2 kernels per step ----
  for (int t = 0; t < T_; t++) {
    __bf16* Xcur  = (t & 1) ? Xc1 : Xc0;
    __bf16* Xnext = (t & 1) ? Xc0 : Xc1;
    gates_lstm<<<K_ / 16, 512, 0, stream>>>(Xcur, Wcomb, b_ih, b_hh, cs, Xnext, hs_bf);
    attn_fused<<<B_, 512, 0, stream>>>(h, hp_bf, hs_bf, phi_bf, phi_b, cd_bf, cd_b, y,
                                       logps, atts, Xnext, t);
  }
}
